// DLGPE_283467841742
// MI455X (gfx1250) — matrix operand order repaired and checked by reading, not compiled, not run on hardware
//
#include <hip/hip_runtime.h>
#include <math.h>

#define WSZ 8
#define HEADS 4
#define DHEAD 48
#define SHIFT 3
#define DIM 192
#define BB 8
#define HH 128
#define WW 128
#define HWSZ (HH*WW)
#define EPSLN 1e-6f

typedef __attribute__((ext_vector_type(16))) _Float16 v16h;
typedef __attribute__((ext_vector_type(8)))  float    v8f;
typedef __attribute__((ext_vector_type(4)))  unsigned int v4u;
typedef __attribute__((ext_vector_type(8)))  int v8i;
typedef __attribute__((ext_vector_type(4)))  int v4i;

#if __has_builtin(__builtin_amdgcn_tensor_load_to_lds)
#define USE_TDM 1
#warning "CDNA5 probe: tensor_load_to_lds builtin AVAILABLE -> TDM V-tile path enabled"
#else
#define USE_TDM 0
#warning "CDNA5 probe: tensor_load_to_lds builtin MISSING -> scalar V-load fallback"
#endif

// ---------------------------------------------------------------------------
// WMMA helpers (CDNA5 v_wmma_f32_16x16x32_f16, wave32)
// A-fragment: lane l<16 -> row M=l, K halves {kb+0..7, kb+16..23};
//             lane l>=16 -> row M=l-16, K halves {kb+8..15, kb+24..31}
// B-fragment: lane l<16 -> col N=l, K halves {kb+0..7, kb+16..23};
//             lane l>=16 -> col N=l-16, K halves {kb+8..15, kb+24..31}
// C/D:        element (M = r + 8*(lane>=16), N = lane&15) in VGPR r
// ---------------------------------------------------------------------------
__device__ __forceinline__ v8f wmma16(v16h a, v16h b, v8f c) {
  return __builtin_amdgcn_wmma_f32_16x16x32_f16(false, a, false, b, (short)0, c,
                                                false, false);
}

__device__ __forceinline__ v16h frag_a(const _Float16* __restrict__ M, int S,
                                       int row, int kb, int lane) {
  const _Float16* p = M + (size_t)row * S + kb + ((lane & 16) ? 8 : 0);
  v16h r;
  reinterpret_cast<uint4*>(&r)[0] = reinterpret_cast<const uint4*>(p)[0];
  reinterpret_cast<uint4*>(&r)[1] = reinterpret_cast<const uint4*>(p + 16)[0];
  return r;
}
__device__ __forceinline__ v16h frag_b(const _Float16* __restrict__ M, int S,
                                       int col, int kb, int lane) {
  const _Float16* p = M + (size_t)col * S + kb + ((lane & 16) ? 8 : 0);
  v16h r;
  reinterpret_cast<uint4*>(&r)[0] = reinterpret_cast<const uint4*>(p)[0];
  reinterpret_cast<uint4*>(&r)[1] = reinterpret_cast<const uint4*>(p + 16)[0];
  return r;
}

__device__ __forceinline__ float gelu_exact(float x) {
  return 0.5f * x * (1.0f + erff(x * 0.70710678118654752f));
}

#if USE_TDM
// ---------------------------------------------------------------------------
// TDM: async-load one f16 tile (8 px x 8 rows x 48 channels) global -> LDS.
// D# packing per CDNA5 ISA 8.3/8.4: count=1, type=2(image), data_size=1 (2B),
// tile dims (8,8,48), tensor dims (8,8,48), strides (WW, HWSZ). LDS receives
// the tile in [z][y][x] order == Vs[d][m] (m = y*8+x), matching the WMMA
// B-fragment layout directly. Wave-level op; issue from a uniform branch.
// ---------------------------------------------------------------------------
__device__ __forceinline__ void tdm_load_v_tile(const _Float16* gsrc, void* ldst) {
  unsigned long long ga = (unsigned long long)(uintptr_t)gsrc;  // byte address
  unsigned int laddr = (unsigned int)(uintptr_t)ldst;           // LDS byte offset
  v4u g0 = { 1u,                                   // count=1, user mode
             laddr,                                // lds_addr
             (unsigned int)ga,                     // global_addr[31:0]
             ((unsigned int)(ga >> 32) & 0x01FFFFFFu) | (2u << 30) };  // type=2
  v8i g1 = { (int)(1u << 16),                      // data_size=1 (2 bytes)
             (int)(8u << 16),                      // tensor_dim0 lo16 = 8
             (int)(8u << 16),                      // dim0 hi=0 | tensor_dim1 lo16 = 8
             (int)(8u << 16),                      // dim1 hi=0 | tile_dim0 = 8
             (int)(8u | (48u << 16)),              // tile_dim1=8, tile_dim2=48
             (int)WW,                              // tensor_dim0_stride lo32
             (int)((unsigned)HWSZ << 16),          // s0 hi=0 | tensor_dim1_stride lo16
             0 };                                  // tensor_dim1_stride hi
  v4i g2 = { 48, 0, 0, 0 };                        // tensor_dim2=48, tile_dim3 unused
  v4i g3 = { 0, 0, 0, 0 };                         // dims 3/4 unused
#if __clang_major__ >= 23
  v8i g4 = { 0, 0, 0, 0, 0, 0, 0, 0 };
  __builtin_amdgcn_tensor_load_to_lds(g0, g1, g2, g3, g4, 0);
#else
  __builtin_amdgcn_tensor_load_to_lds(g0, g1, g2, g3, 0);
#endif
}

__device__ __forceinline__ void tdm_wait0() {
#if __has_builtin(__builtin_amdgcn_s_wait_tensorcnt)
  __builtin_amdgcn_s_wait_tensorcnt((short)0);
#else
  asm volatile("s_wait_tensorcnt 0x0" ::: "memory");
#endif
}
__device__ __forceinline__ void tdm_wait1() {
#if __has_builtin(__builtin_amdgcn_s_wait_tensorcnt)
  __builtin_amdgcn_s_wait_tensorcnt((short)1);
#else
  asm volatile("s_wait_tensorcnt 0x1" ::: "memory");
#endif
}
#endif  // USE_TDM

// ---------------------------------------------------------------------------
// global mean pool: one block per (b, c)
// ---------------------------------------------------------------------------
__global__ void k_pool(const float* __restrict__ x, float* __restrict__ pooled) {
  const float* p = x + (size_t)blockIdx.x * HWSZ;
  float s = 0.f;
  for (int i = threadIdx.x; i < HWSZ; i += 256) s += p[i];
  __shared__ float red[256];
  red[threadIdx.x] = s;
  __syncthreads();
  for (int st = 128; st > 0; st >>= 1) {
    if ((int)threadIdx.x < st) red[threadIdx.x] += red[threadIdx.x + st];
    __syncthreads();
  }
  if (threadIdx.x == 0) pooled[blockIdx.x] = red[0] * (1.0f / HWSZ);
}

// ---------------------------------------------------------------------------
// channel-attention gate MLP -> cw[B][3]
// ---------------------------------------------------------------------------
__global__ void k_gate(const float* __restrict__ pooled,
                       const float* w1, const float* b1, const float* g1, const float* be1,
                       const float* w2, const float* b2, const float* g2, const float* be2,
                       float* __restrict__ cw) {
  __shared__ float hs[48], as[48];
  int t = threadIdx.x;
  for (int b = 0; b < BB; ++b) {
    const float* pb = pooled + b * DIM;
    if (t < 48) {
      float acc = b1[t];
      for (int c = 0; c < DIM; ++c) acc += w1[t * DIM + c] * pb[c];
      hs[t] = acc;
    }
    __syncthreads();
    if (t == 0) {
      float mu = 0.f;
      for (int j = 0; j < 48; ++j) mu += hs[j];
      mu *= (1.f / 48.f);
      float var = 0.f;
      for (int j = 0; j < 48; ++j) { float d = hs[j] - mu; var += d * d; }
      var *= (1.f / 48.f);
      float rs = rsqrtf(var + EPSLN);
      for (int j = 0; j < 48; ++j)
        as[j] = gelu_exact((hs[j] - mu) * rs * g1[j] + be1[j]);
      float z[3];
      for (int k = 0; k < 3; ++k) {
        float a = b2[k];
        for (int j = 0; j < 48; ++j) a += w2[k * 48 + j] * as[j];
        z[k] = a;
      }
      float m2 = (z[0] + z[1] + z[2]) * (1.f / 3.f);
      float v2 = 0.f;
      for (int k = 0; k < 3; ++k) { float d = z[k] - m2; v2 += d * d; }
      v2 *= (1.f / 3.f);
      float rs2 = rsqrtf(v2 + EPSLN);
      float y[3], mx = -1e30f;
      for (int k = 0; k < 3; ++k) {
        y[k] = (z[k] - m2) * rs2 * g2[k] + be2[k];
        mx = fmaxf(mx, y[k]);
      }
      float s = 0.f;
      for (int k = 0; k < 3; ++k) { y[k] = __expf(y[k] - mx); s += y[k]; }
      for (int k = 0; k < 3; ++k) cw[b * 3 + k] = 0.25f + 0.25f * (y[k] / s);
    }
    __syncthreads();
  }
}

// ---------------------------------------------------------------------------
// weight repacks (fp32 -> f16)
// ---------------------------------------------------------------------------
__global__ void k_cvt_attnw(const float* __restrict__ w0, const float* __restrict__ w1,
                            _Float16* __restrict__ dst) {
  const int n = 3 * DIM * DIM;
  for (int i = blockIdx.x * blockDim.x + threadIdx.x; i < 2 * n;
       i += gridDim.x * blockDim.x)
    dst[i] = (_Float16)((i < n) ? w0[i] : w1[i - n]);
}

__global__ void k_cvt_convw(const float* __restrict__ w, _Float16* __restrict__ dst,
                            int ksize, int Kpad) {
  int kk = ksize * ksize;
  int Ktot = 48 * kk;
  for (int i = blockIdx.x * blockDim.x + threadIdx.x; i < 48 * Kpad;
       i += gridDim.x * blockDim.x) {
    int co = i / Kpad, k = i % Kpad;
    float v = (k < Ktot) ? w[co * Ktot + k] : 0.f;
    dst[i] = (_Float16)v;
  }
}

// ---------------------------------------------------------------------------
// relative-position bias table [4][64][64]
// ---------------------------------------------------------------------------
__global__ void k_bias(const float* __restrict__ rpb, float* __restrict__ biasT) {
  for (int i = blockIdx.x * blockDim.x + threadIdx.x; i < HEADS * 64 * 64;
       i += gridDim.x * blockDim.x) {
    int h = i >> 12, rem = i & 4095, n = rem >> 6, m = rem & 63;
    int idx = ((n >> 3) - (m >> 3) + 7) * 15 + ((n & 7) - (m & 7) + 7);
    biasT[i] = rpb[idx * HEADS + h];
  }
}

// ---------------------------------------------------------------------------
// QKV 1x1 conv as WMMA GEMM [576x192] x [192x64px] + q-LayerNorm epilogue
// out: qkvh[b][576][HW] f16 (q already LayerNormed)
// ---------------------------------------------------------------------------
__global__ void __launch_bounds__(256)
k_qkv(const float* __restrict__ x, const _Float16* __restrict__ wh,
      const float* __restrict__ bias, const float* __restrict__ lng,
      const float* __restrict__ lnb, const float* __restrict__ cw, int gidx,
      _Float16* __restrict__ qkvh) {
  const int XK = 200;  // padded K stride (halves), 400B = 16B aligned
  __shared__ _Float16 Xs[64 * XK];
  __shared__ _Float16 Qs[64 * DIM];
  __shared__ float rsum[4 * 64], rsq[4 * 64];

  int b = blockIdx.x >> 8;
  int p0 = (blockIdx.x & 255) * 64;
  const float gate = cw[b * 3 + gidx];
  const float* xb = x + (size_t)b * DIM * HWSZ;
  _Float16* qb = qkvh + (size_t)b * 3 * DIM * HWSZ;

  for (int i = threadIdx.x; i < DIM * 64; i += 256) {
    int ci = i >> 6, n = i & 63;
    Xs[n * XK + ci] = (_Float16)(xb[(size_t)ci * HWSZ + p0 + n] * gate);
  }
  __syncthreads();

  int wave = threadIdx.x >> 5, lane = threadIdx.x & 31;
  for (int m = wave; m < 36; m += 8) {
    if (m + 8 < 36)  // warm L2/L0 for the next weight M-tile (global_prefetch_b8)
      __builtin_prefetch(wh + (size_t)(m + 8) * 16 * DIM, 0, 1);
    v8f acc[4] = {};
    for (int kc = 0; kc < 6; ++kc) {
      v16h a = frag_a(wh, DIM, m * 16 + (lane & 15), kc * 32, lane);
#pragma unroll
      for (int nt = 0; nt < 4; ++nt) {
        v16h bf = frag_b(Xs, XK, nt * 16 + (lane & 15), kc * 32, lane);
        acc[nt] = wmma16(a, bf, acc[nt]);
      }
    }
#pragma unroll
    for (int nt = 0; nt < 4; ++nt)
#pragma unroll
      for (int r = 0; r < 8; ++r) {
        int ch = m * 16 + r + ((lane & 16) ? 8 : 0);
        int n = nt * 16 + (lane & 15);
        float v = acc[nt][r] + bias[ch];
        if (ch < DIM)
          Qs[n * DIM + ch] = (_Float16)v;          // q -> LN below
        else
          qb[(size_t)ch * HWSZ + p0 + n] = (_Float16)v;  // k, v
      }
  }
  __syncthreads();

  // LayerNorm(q) over 192 channels per pixel (4 threads per pixel)
  int n = threadIdx.x & 63, part = threadIdx.x >> 6;
  float s = 0.f, s2 = 0.f;
  for (int c = part * 48; c < part * 48 + 48; ++c) {
    float v = (float)Qs[n * DIM + c];
    s += v; s2 += v * v;
  }
  rsum[part * 64 + n] = s;
  rsq[part * 64 + n] = s2;
  __syncthreads();
  float ts = rsum[n] + rsum[64 + n] + rsum[128 + n] + rsum[192 + n];
  float ts2 = rsq[n] + rsq[64 + n] + rsq[128 + n] + rsq[192 + n];
  float mu = ts * (1.f / DIM);
  float rs = rsqrtf(ts2 * (1.f / DIM) - mu * mu + EPSLN);
  for (int c = part * 48; c < part * 48 + 48; ++c) {
    float v = ((float)Qs[n * DIM + c] - mu) * rs * lng[c] + lnb[c];
    qb[(size_t)c * HWSZ + p0 + n] = (_Float16)v;
  }
}

// ---------------------------------------------------------------------------
// window attention: one WG per 8x8 window, one wave per 16-row block,
// heads looped. q gathered (and out scattered) at +shift; k/v unshifted.
// V tiles stream in via the Tensor Data Mover, double-buffered so the DMA for
// head h+1 overlaps head h's WMMA work (TENSORcnt ordered waits).
// ---------------------------------------------------------------------------
__global__ void __launch_bounds__(128)
k_attn(const _Float16* __restrict__ qkvh, const float* __restrict__ biasT,
       float* __restrict__ out, int shift) {
  __shared__ _Float16 Qs[64 * 64];     // [n][d] (d padded to 64)
  __shared__ _Float16 Ks[64 * 64];     // [m][d]
  __shared__ _Float16 Vs[2][48 * 64];  // [d][m]  (TDM tile order [z][y][x])
  __shared__ _Float16 As[64 * 80];     // softmaxed attn [n][m], stride 160B

  int b = blockIdx.x >> 8;
  int win = blockIdx.x & 255;
  int r0 = (win >> 4) * WSZ, c0 = (win & 15) * WSZ;
  const _Float16* qkb = qkvh + (size_t)b * 3 * DIM * HWSZ;
  float* ob = out + (size_t)b * DIM * HWSZ;
  const float scale = 0.14433756729740643f;  // 1/sqrt(48)
  int lane = threadIdx.x & 31, wave = threadIdx.x >> 5;

#if USE_TDM
  if (wave == 0)  // wave-uniform branch: prologue TDM issue for head 0
    tdm_load_v_tile(qkb + (size_t)(2 * DIM) * HWSZ + r0 * WW + c0, (void*)Vs[0]);
#endif

  for (int h = 0; h < HEADS; ++h) {
    __syncthreads();
#if USE_TDM
    // pipeline: kick off head h+1's V tile while we compute head h
    if (wave == 0 && h + 1 < HEADS)
      tdm_load_v_tile(qkb + (size_t)(2 * DIM + (h + 1) * DHEAD) * HWSZ + r0 * WW + c0,
                      (void*)Vs[(h + 1) & 1]);
#endif
    for (int i = threadIdx.x; i < 64 * 64; i += 128) {
      int d = i >> 6, n = i & 63;
      _Float16 qv = (_Float16)0.f, kv = (_Float16)0.f;
      if (d < DHEAD) {
        int rq = (r0 + (n >> 3) + shift) & (HH - 1);
        int cq = (c0 + (n & 7) + shift) & (WW - 1);
        qv = qkb[(size_t)(h * DHEAD + d) * HWSZ + rq * WW + cq];
        kv = qkb[(size_t)(DIM + h * DHEAD + d) * HWSZ + (r0 + (n >> 3)) * WW + c0 + (n & 7)];
      }
      Qs[n * 64 + d] = qv;
      Ks[n * 64 + d] = kv;
    }
#if USE_TDM
    if (wave == 0) {
      // TENSORcnt decrements in order: <=1 with two outstanding means head h done
      if (h + 1 < HEADS) tdm_wait1(); else tdm_wait0();
    }
#else
    for (int i = threadIdx.x; i < DHEAD * 64; i += 128) {
      int d = i >> 6, m = i & 63;
      Vs[h & 1][d * 64 + m] =
          qkb[(size_t)(2 * DIM + h * DHEAD + d) * HWSZ + (r0 + (m >> 3)) * WW + c0 + (m & 7)];
    }
#endif
    __syncthreads();

    // scores = Q K^T (K padded 48->64 with zeros)
    v8f sc[4] = {};
    for (int kc = 0; kc < 2; ++kc) {
      v16h a = frag_a(Qs, 64, wave * 16 + (lane & 15), kc * 32, lane);
#pragma unroll
      for (int mt = 0; mt < 4; ++mt) {
        v16h bf = frag_b(Ks, 64, mt * 16 + (lane & 15), kc * 32, lane);
        sc[mt] = wmma16(a, bf, sc[mt]);
      }
    }
    // scale + relative position bias
    const float* bh = biasT + h * 4096;
#pragma unroll
    for (int mt = 0; mt < 4; ++mt)
#pragma unroll
      for (int r = 0; r < 8; ++r) {
        int n = wave * 16 + r + ((lane & 16) ? 8 : 0);
        sc[mt][r] = sc[mt][r] * scale + bh[n * 64 + mt * 16 + (lane & 15)];
      }
    // softmax over m: rows live in 16-lane halves -> xor-shuffle reduce
#pragma unroll
    for (int r = 0; r < 8; ++r) {
      float mx = fmaxf(fmaxf(sc[0][r], sc[1][r]), fmaxf(sc[2][r], sc[3][r]));
      for (int o = 8; o >= 1; o >>= 1) mx = fmaxf(mx, __shfl_xor(mx, o, 32));
      float sum = 0.f;
#pragma unroll
      for (int mt = 0; mt < 4; ++mt) {
        sc[mt][r] = __expf(sc[mt][r] - mx);
        sum += sc[mt][r];
      }
      for (int o = 8; o >= 1; o >>= 1) sum += __shfl_xor(sum, o, 32);
      float inv = 1.f / sum;
#pragma unroll
      for (int mt = 0; mt < 4; ++mt) sc[mt][r] *= inv;
    }
    // relayout C -> A through LDS (each wave touches only its own rows)
#pragma unroll
    for (int mt = 0; mt < 4; ++mt)
#pragma unroll
      for (int r = 0; r < 8; ++r) {
        int n = wave * 16 + r + ((lane & 16) ? 8 : 0);
        As[n * 80 + mt * 16 + (lane & 15)] = (_Float16)sc[mt][r];
      }
    // out^T[n][d] = attn x V^T
    v8f ov[3] = {};
    for (int kc = 0; kc < 2; ++kc) {
      v16h a = frag_a(As, 80, wave * 16 + (lane & 15), kc * 32, lane);
#pragma unroll
      for (int dt = 0; dt < 3; ++dt) {
        v16h bf = frag_b(Vs[h & 1], 64, dt * 16 + (lane & 15), kc * 32, lane);
        ov[dt] = wmma16(a, bf, ov[dt]);
      }
    }
    // accumulate into output at shifted positions
#pragma unroll
    for (int dt = 0; dt < 3; ++dt)
#pragma unroll
      for (int r = 0; r < 8; ++r) {
        int n = wave * 16 + r + ((lane & 16) ? 8 : 0);
        int rr = (r0 + (n >> 3) + shift) & (HH - 1);
        int cc = (c0 + (n & 7) + shift) & (WW - 1);
        size_t idx = (size_t)(h * DHEAD + dt * 16 + (lane & 15)) * HWSZ + rr * WW + cc;
        ob[idx] += ov[dt][r];
      }
  }
}

// ---------------------------------------------------------------------------
// conv (3x3 or 1x1, 48->48) as implicit-GEMM WMMA + LN + GELU epilogue.
// One WG = 32 contiguous pixels in one image row, all 48 out channels.
// ---------------------------------------------------------------------------
__global__ void __launch_bounds__(128)
k_conv(const float* __restrict__ in, int inC, int inOff,
       const float* __restrict__ cw, int gidx, const _Float16* __restrict__ wt,
       int Kpad, int ksize, const float* __restrict__ bias,
       const float* __restrict__ lng, const float* __restrict__ lnb,
       float* __restrict__ outp, int outC, int outOff) {
  const int XK = 456;  // stride halves, 912B = 16B aligned
  __shared__ _Float16 Xs[32 * XK];
  __shared__ float Os[32 * 52];

  int b = blockIdx.x >> 9;
  int tile = blockIdx.x & 511;
  int row = tile >> 2;
  int c0 = (tile & 3) * 32;
  float gate = (gidx >= 0) ? cw[b * 3 + gidx] : 1.0f;
  const float* ib = in + ((size_t)b * inC + inOff) * HWSZ;
  int Ktot = 48 * ksize * ksize;

  __builtin_prefetch(wt, 0, 1);  // weights are L2-hot; warm near cache

  // zero-fill K pad region
  for (int i = threadIdx.x; i < 32 * (Kpad - Ktot); i += 128) {
    int n = i / (Kpad - Ktot);
    Xs[n * XK + Ktot + i % (Kpad - Ktot)] = (_Float16)0.f;
  }
  if (ksize == 3) {
    for (int i = threadIdx.x; i < 48 * 3 * 34; i += 128) {
      int ci = i / 102, rem = i % 102;
      int dy = rem / 34 - 1, cs = rem % 34 - 1;
      int rr = row + dy, cc = c0 + cs;
      float v = 0.f;
      if ((unsigned)rr < (unsigned)HH && (unsigned)cc < (unsigned)WW)
        v = ib[(size_t)ci * HWSZ + rr * WW + cc] * gate;
      _Float16 hv = (_Float16)v;
#pragma unroll
      for (int dx = -1; dx <= 1; ++dx) {
        int n = cs - dx;
        if ((unsigned)n < 32u)
          Xs[n * XK + ci * 9 + (dy + 1) * 3 + (dx + 1)] = hv;
      }
    }
  } else {
    for (int i = threadIdx.x; i < 48 * 32; i += 128) {
      int ci = i >> 5, n = i & 31;
      Xs[n * XK + ci] = (_Float16)(ib[(size_t)ci * HWSZ + row * WW + c0 + n] * gate);
    }
  }
  __syncthreads();

  int wave = threadIdx.x >> 5, lane = threadIdx.x & 31;
  int Kch = Kpad / 32;
  for (int job = wave; job < 6; job += 4) {
    int mt = job >> 1, nt = job & 1;
    v8f acc = {};
    for (int kc = 0; kc < Kch; ++kc) {
      v16h a = frag_a(wt, Kpad, mt * 16 + (lane & 15), kc * 32, lane);
      v16h bf = frag_b(Xs, XK, nt * 16 + (lane & 15), kc * 32, lane);
      acc = wmma16(a, bf, acc);
    }
#pragma unroll
    for (int r = 0; r < 8; ++r) {
      int co = mt * 16 + r + ((lane & 16) ? 8 : 0);
      Os[(nt * 16 + (lane & 15)) * 52 + co] = acc[r] + bias[co];
    }
  }
  __syncthreads();

  if (threadIdx.x < 32) {
    int n = threadIdx.x;
    float s = 0.f, s2 = 0.f;
    for (int c = 0; c < 48; ++c) {
      float v = Os[n * 52 + c];
      s += v; s2 += v * v;
    }
    float mu = s * (1.f / 48.f);
    float rs = rsqrtf(s2 * (1.f / 48.f) - mu * mu + EPSLN);
    float* obp = outp + ((size_t)b * outC + outOff) * HWSZ + row * WW + c0;
    for (int c = 0; c < 48; ++c) {
      float v = (Os[n * 52 + c] - mu) * rs * lng[c] + lnb[c];
      obp[(size_t)c * HWSZ + n] = gelu_exact(v);
    }
  }
}

// ---------------------------------------------------------------------------
extern "C" void kernel_launch(void* const* d_in, const int* in_sizes, int n_in,
                              void* d_out, int out_size, void* d_ws, size_t ws_size,
                              hipStream_t stream) {
  (void)in_sizes; (void)n_in; (void)out_size; (void)ws_size;
  // input leaves (setup_inputs insertion order, nested dicts/lists depth-first)
  const float* x       = (const float*)d_in[0];
  const float* ca1_w   = (const float*)d_in[1];
  const float* ca1_b   = (const float*)d_in[2];
  const float* ca1_g   = (const float*)d_in[3];
  const float* ca1_be  = (const float*)d_in[4];
  const float* ca2_w   = (const float*)d_in[5];
  const float* ca2_b   = (const float*)d_in[6];
  const float* ca2_g   = (const float*)d_in[7];
  const float* ca2_be  = (const float*)d_in[8];
  const float* wqkv_w  = (const float*)d_in[9];
  const float* wqkv_b  = (const float*)d_in[10];
  const float* swqkv_w = (const float*)d_in[11];
  const float* swqkv_b = (const float*)d_in[12];
  const float* norm_g  = (const float*)d_in[13];
  const float* norm_b  = (const float*)d_in[14];
  const float* rpb     = (const float*)d_in[15];
  // convs: 7 layers x (w,b,g,be) starting at leaf 16
  auto cvw  = [&](int l) { return (const float*)d_in[16 + l * 4 + 0]; };
  auto cvb  = [&](int l) { return (const float*)d_in[16 + l * 4 + 1]; };
  auto cvg  = [&](int l) { return (const float*)d_in[16 + l * 4 + 2]; };
  auto cvbe = [&](int l) { return (const float*)d_in[16 + l * 4 + 3]; };

  char* ws = (char*)d_ws;
  size_t off = 0;
  auto alloc = [&](size_t bytes) -> void* {
    off = (off + 255) & ~(size_t)255;
    void* p = ws + off;
    off += bytes;
    return p;
  };
  float*     pooled = (float*)alloc((size_t)BB * DIM * 4);
  float*     cw     = (float*)alloc((size_t)BB * 3 * 4);
  _Float16*  whA    = (_Float16*)alloc((size_t)2 * 3 * DIM * DIM * 2);
  _Float16*  wc[7];
  for (int l = 0; l < 6; ++l) wc[l] = (_Float16*)alloc((size_t)48 * 448 * 2);
  wc[6] = (_Float16*)alloc((size_t)48 * 64 * 2);
  float*     biasT  = (float*)alloc((size_t)HEADS * 64 * 64 * 4);
  _Float16*  qkvh   = (_Float16*)alloc((size_t)BB * 3 * DIM * HWSZ * 2);
  // conv ping/pong buffers alias qkv scratch (used at disjoint times)
  float* t0 = (float*)qkvh;
  float* t1 = t0 + (size_t)BB * 48 * HWSZ;

  float* o = (float*)d_out;

  // stage 0: gates + weight repacks + bias table
  k_pool<<<BB * DIM, 256, 0, stream>>>(x, pooled);
  k_gate<<<1, 64, 0, stream>>>(pooled, ca1_w, ca1_b, ca1_g, ca1_be,
                               ca2_w, ca2_b, ca2_g, ca2_be, cw);
  k_cvt_attnw<<<256, 256, 0, stream>>>(wqkv_w, swqkv_w, whA);
  for (int l = 0; l < 7; ++l) {
    int ks = (l == 6) ? 1 : 3, Kpad = (l == 6) ? 64 : 448;
    k_cvt_convw<<<96, 256, 0, stream>>>(cvw(l), wc[l], ks, Kpad);
  }
  k_bias<<<64, 256, 0, stream>>>(rpb, biasT);

  // stage 1: multi-kernel conv branch writes d_out (all 192 channels, '=')
  const int CG = BB * HH * (WW / 32);
  k_conv<<<CG, 128, 0, stream>>>(x, DIM, 0,   cw, 0,  wc[0], 448, 3, cvb(0), cvg(0), cvbe(0), t0, 48, 0);
  k_conv<<<CG, 128, 0, stream>>>(t0, 48, 0,   cw, -1, wc[1], 448, 3, cvb(1), cvg(1), cvbe(1), t1, 48, 0);
  k_conv<<<CG, 128, 0, stream>>>(t1, 48, 0,   cw, -1, wc[2], 448, 3, cvb(2), cvg(2), cvbe(2), o, DIM, 0);
  k_conv<<<CG, 128, 0, stream>>>(x, DIM, 48,  cw, 0,  wc[3], 448, 3, cvb(3), cvg(3), cvbe(3), t0, 48, 0);
  k_conv<<<CG, 128, 0, stream>>>(t0, 48, 0,   cw, -1, wc[4], 448, 3, cvb(4), cvg(4), cvbe(4), o, DIM, 48);
  k_conv<<<CG, 128, 0, stream>>>(x, DIM, 96,  cw, 0,  wc[5], 448, 3, cvb(5), cvg(5), cvbe(5), o, DIM, 96);
  k_conv<<<CG, 128, 0, stream>>>(x, DIM, 144, cw, 0,  wc[6], 64,  1, cvb(6), cvg(6), cvbe(6), o, DIM, 144);

  // stage 2: window attention branch (accumulates into d_out)
  k_qkv<<<BB * 256, 256, 0, stream>>>(x, whA, wqkv_b, norm_g, norm_b, cw, 1, qkvh);
  k_attn<<<BB * 256, 128, 0, stream>>>(qkvh, biasT, o, 0);

  // stage 3: shifted window attention branch (accumulates into d_out)
  k_qkv<<<BB * 256, 256, 0, stream>>>(x, whA + 3 * DIM * DIM, swqkv_b, norm_g,
                                      norm_b, cw, 2, qkvh);
  k_attn<<<BB * 256, 128, 0, stream>>>(qkvh, biasT, o, SHIFT);
}
